// SelfAttention_56349970923693
// MI455X (gfx1250) — matrix operand order repaired and checked by reading, not compiled, not run on hardware
//
#include <hip/hip_runtime.h>
#include <hip/hip_bf16.h>
#include <cstdint>

// ---------------- problem constants (from reference) ----------------
#define B_    4
#define T_    2048
#define D_    1024
#define NWAVE 8
#define WCOLS (D_ / NWAVE)      // 128 output columns per wave
#define CTILES (WCOLS / 16)     // 8 accumulator tiles per wave (per row tile)
#define KTILE 128               // keys per outer iteration

typedef __attribute__((ext_vector_type(16))) __bf16          v16bf;
typedef __attribute__((ext_vector_type(8)))  float           v8f;
typedef __attribute__((ext_vector_type(4)))  float           v4f;
typedef __attribute__((ext_vector_type(2)))  float           v2f;
typedef __attribute__((ext_vector_type(8)))  unsigned short  v8us;

union AB {
    v16bf          bf;
    v8us           u[2];
    unsigned       u32[8];
    unsigned short us[16];
};

// pack two fp32 into one dword of two bf16 (truncation) with a single v_perm_b32
__device__ __forceinline__ unsigned pk2bf(float lo, float hi) {
    return __builtin_amdgcn_perm(__builtin_bit_cast(unsigned, hi),
                                 __builtin_bit_cast(unsigned, lo),
                                 0x07060302u);
}
__device__ __forceinline__ unsigned short f2bf(float f) {      // truncate
    return (unsigned short)(__builtin_bit_cast(unsigned, f) >> 16);
}

// =====================================================================
// Prep kernel 1: K fp32 -> bf16, same [b][t][d] layout
// =====================================================================
__global__ __launch_bounds__(256)
void convert_k_bf16(const float* __restrict__ K, unsigned* __restrict__ Kbf32) {
    size_t total = (size_t)B_ * T_ * D_ / 2;
    for (size_t i = (size_t)blockIdx.x * blockDim.x + threadIdx.x; i < total;
         i += (size_t)gridDim.x * blockDim.x) {
        v2f f = *(const v2f*)(K + 2 * i);
        Kbf32[i] = pk2bf(f.x, f.y);
    }
}

// =====================================================================
// Prep kernel 2: V fp32 [b][t][d] -> bf16 transposed Vtg [b][d][t]
// =====================================================================
__global__ __launch_bounds__(256)
void transpose_v_bf16(const float* __restrict__ V, unsigned short* __restrict__ Vtg) {
    __shared__ unsigned short tile[32][33];
    const int blk = blockIdx.x;
    const int b   = blk >> 11;            // (T_/32)*(D_/32) = 2048 tiles per batch
    const int rem = blk & 2047;
    const int k0  = (rem >> 5) << 5;
    const int d0  = (rem & 31) << 5;
    const int tx  = threadIdx.x & 31;
    const int ty  = threadIdx.x >> 5;
    #pragma unroll
    for (int i = 0; i < 4; ++i) {
        int kk = ty + (i << 3);
        tile[kk][tx] = f2bf(V[((size_t)(b * T_) + k0 + kk) * D_ + d0 + tx]);
    }
    __syncthreads();
    #pragma unroll
    for (int i = 0; i < 4; ++i) {
        int dd = ty + (i << 3);
        Vtg[((size_t)(b * D_) + d0 + dd) * T_ + k0 + tx] = tile[tx][dd];
    }
}

// =====================================================================
// Main kernel (workspace path): 32 query rows / workgroup, bf16 K from
// global, bf16 V^T async-DMA'd into double-buffered LDS.
// =====================================================================
#define QS_PITCH 1032
#define PB_PITCH 136
#define VC_PITCH 80                                    // bytes per V column in LDS (64 used)
#define QS2_BYTES (32 * QS_PITCH * 2)                  // 66048
#define PB2_OFF   QS2_BYTES
#define WMAX2_OFF (PB2_OFF + 32 * PB_PITCH * 2)
#define WSUM2_OFF (WMAX2_OFF + NWAVE * 32 * 4)
#define ROWM2_OFF (WSUM2_OFF + NWAVE * 32 * 4)
#define ROWL2_OFF (ROWM2_OFF + 32 * 4)
#define ROWF2_OFF (ROWL2_OFF + 32 * 4)
#define VTB_OFF   (ROWF2_OFF + 32 * 4)                 // 77184 (16B aligned)
#define VTB_STRIDE (D_ * VC_PITCH)                     // 81920 per buffer
#define SMEM2_BYTES (VTB_OFF + 2 * VTB_STRIDE)         // 241024 (~235KB, WGP has 320KB)

// async-copy this wave's 128-col x 32-key V^T slab (8KB) into LDS:
// 16 x global_load_async_to_lds_b128, no VGPR data path, tracked by ASYNCcnt
__device__ __forceinline__ void issue_vtile(const unsigned short* __restrict__ Vtg,
                                            char* vtbuf_wave,   // LDS base incl. wave offset
                                            int b, int w, int l, int ksc) {
    const char* gbase = (const char*)(Vtg + ((size_t)(b * D_) + w * WCOLS) * T_ + ksc);
    #pragma unroll
    for (int j = 0; j < 16; ++j) {
        int cid  = j * 32 + l;          // 512 16B-chunks per slab
        int col  = cid >> 2;            // 0..127 (local column)
        int part = cid & 3;             // 4 x 16B per 64B column
        const char* g = gbase + (size_t)col * (T_ * 2) + part * 16;
        unsigned loff = (unsigned)(uintptr_t)(vtbuf_wave + col * VC_PITCH + part * 16);
        asm volatile("global_load_async_to_lds_b128 %0, %1, off"
                     :: "v"(loff), "v"(g) : "memory");
    }
}

__global__ __launch_bounds__(256)
void flash_attn_ws(const float* __restrict__ Q,
                   const unsigned short* __restrict__ Kbf,
                   const unsigned short* __restrict__ Vtg,
                   const int* __restrict__ AM,
                   float* __restrict__ O)
{
    extern __shared__ char smem[];
    unsigned short* Qs = (unsigned short*)smem;                 // [32][QS_PITCH] bf16, pre-scaled
    unsigned short* Pb = (unsigned short*)(smem + PB2_OFF);     // [32][PB_PITCH] bf16 probabilities
    float* wmax = (float*)(smem + WMAX2_OFF);
    float* wsum = (float*)(smem + WSUM2_OFF);
    float* rowm = (float*)(smem + ROWM2_OFF);
    float* rowl = (float*)(smem + ROWL2_OFF);
    float* rowf = (float*)(smem + ROWF2_OFF);

    const int tid  = threadIdx.x;
    const int w    = tid >> 5;
    const int l    = tid & 31;
    const int half = l >> 4;
    const int n    = l & 15;

    char* vt0 = smem + VTB_OFF + (size_t)(w * WCOLS) * VC_PITCH;
    char* vt1 = vt0 + VTB_STRIDE;

    const int b  = blockIdx.x >> 6;          // 64 query tiles per batch
    const int q0 = (blockIdx.x & 63) << 5;

    const float scale = 0.03125f;            // 1/sqrt(1024)

    // ---- stage scaled Q (32 rows) into LDS as bf16 ----
    for (int idx = tid; idx < (32 * D_) / 2; idx += 256) {
        int r  = idx >> 9;
        int d2 = (idx & 511) << 1;
        const float* qp = Q + ((size_t)(b * T_ + q0 + r)) * D_ + d2;
        *(unsigned*)(Qs + r * QS_PITCH + d2) = pk2bf(qp[0] * scale, qp[1] * scale);
    }
    if (tid < 32) { rowm[tid] = -1e30f; rowl[tid] = 0.0f; rowf[tid] = 1.0f; }
    __syncthreads();

    v8f acc[2][CTILES];
    #pragma unroll
    for (int t = 0; t < 2; ++t)
        #pragma unroll
        for (int c = 0; c < CTILES; ++c)
            acc[t][c] = (v8f){0.f,0.f,0.f,0.f,0.f,0.f,0.f,0.f};

    const int kt_count = (q0 + 32 + KTILE - 1) / KTILE;
    for (int kt = 0; kt < kt_count; ++kt) {
        const int k0   = kt * KTILE;
        const int key  = k0 + (w << 4) + n;
        const int keyc = key < T_ ? key : (T_ - 1);

        // ---- kick off async V^T staging for ks=0,32 (hidden under QK^T) ----
        {
            int ks0 = k0      <= (T_ - 32) ? k0      : (T_ - 32);
            int ks1 = k0 + 32 <= (T_ - 32) ? k0 + 32 : (T_ - 32);
            asm volatile("s_wait_dscnt 0" ::: "memory");   // prior reads of buffers done
            issue_vtile(Vtg, vt0, b, w, l, ks0);
            issue_vtile(Vtg, vt1, b, w, l, ks1);
        }

        // -------- scores: two 16x16 tiles share each K B-matrix --------
        v8f S[2];
        S[0] = (v8f){0.f,0.f,0.f,0.f,0.f,0.f,0.f,0.f};
        S[1] = S[0];
        const unsigned short* kp = Kbf + ((size_t)(b * T_) + keyc) * D_ + (half << 4);
        #pragma unroll 2
        for (int dc = 0; dc < D_; dc += 32) {
            AB bm, a0, a1;
            bm.u[0] = *(const v8us*)(kp + dc);
            bm.u[1] = *(const v8us*)(kp + dc + 8);
            const unsigned short* q0p = Qs + n * QS_PITCH + dc + (half << 4);
            const unsigned short* q1p = q0p + 16 * QS_PITCH;
            a0.u[0] = *(const v8us*)(q0p);
            a0.u[1] = *(const v8us*)(q0p + 8);
            a1.u[0] = *(const v8us*)(q1p);
            a1.u[1] = *(const v8us*)(q1p + 8);
            S[0] = __builtin_amdgcn_wmma_f32_16x16x32_bf16(false, a0.bf, false, bm.bf,
                                                           (short)0, S[0], false, false);
            S[1] = __builtin_amdgcn_wmma_f32_16x16x32_bf16(false, a1.bf, false, bm.bf,
                                                           (short)0, S[1], false, false);
        }

        // -------- mask + per-row max --------
        const int am = (key < T_) ? AM[b * T_ + key] : 0;
        float mrow[2][8];
        #pragma unroll
        for (int t = 0; t < 2; ++t) {
            #pragma unroll
            for (int r = 0; r < 8; ++r) {
                const int qg = q0 + t * 16 + r + (half << 3);
                if (!((key <= qg) && (am != 0))) S[t][r] = -1e30f;
                float m = S[t][r];
                m = fmaxf(m, __shfl_xor(m, 1));
                m = fmaxf(m, __shfl_xor(m, 2));
                m = fmaxf(m, __shfl_xor(m, 4));
                m = fmaxf(m, __shfl_xor(m, 8));
                mrow[t][r] = m;
            }
        }
        if (n == 0) {
            #pragma unroll
            for (int t = 0; t < 2; ++t)
                #pragma unroll
                for (int r = 0; r < 8; ++r)
                    wmax[w * 32 + t * 16 + r + (half << 3)] = mrow[t][r];
        }
        __syncthreads();                              // (A)
        if (w == 0) {
            float mo = rowm[l], mn = mo;
            #pragma unroll
            for (int ww = 0; ww < NWAVE; ++ww) mn = fmaxf(mn, wmax[ww * 32 + l]);
            rowm[l] = mn;
            rowf[l] = __expf(mo - mn);
        }
        __syncthreads();                              // (B)

        // -------- P = exp(S - m); sums; rescale accumulators --------
        float srow[2][8];
        #pragma unroll
        for (int t = 0; t < 2; ++t) {
            #pragma unroll
            for (int r = 0; r < 8; ++r) {
                const int rg = t * 16 + r + (half << 3);
                float p = __expf(S[t][r] - rowm[rg]);
                Pb[rg * PB_PITCH + (w << 4) + n] = f2bf(p);
                float s = p;
                s += __shfl_xor(s, 1);
                s += __shfl_xor(s, 2);
                s += __shfl_xor(s, 4);
                s += __shfl_xor(s, 8);
                srow[t][r] = s;
                const float fac = rowf[rg];
                #pragma unroll
                for (int c = 0; c < CTILES; ++c) acc[t][c][r] *= fac;
            }
        }
        if (n == 0) {
            #pragma unroll
            for (int t = 0; t < 2; ++t)
                #pragma unroll
                for (int r = 0; r < 8; ++r)
                    wsum[w * 32 + t * 16 + r + (half << 3)] = srow[t][r];
        }
        __syncthreads();                              // (C): Pb + wsum visible
        if (w == 0) {
            float s = 0.f;
            #pragma unroll
            for (int ww = 0; ww < NWAVE; ++ww) s += wsum[ww * 32 + l];
            rowl[l] = rowl[l] * rowf[l] + s;
        }

        // -------- PV: B from double-buffered LDS (async-staged) --------
        #pragma unroll
        for (int s = 0; s < 4; ++s) {
            const int ks = s * 32;
            if (s < 3) asm volatile("s_wait_asynccnt 16" ::: "memory");  // older buffer landed
            else       asm volatile("s_wait_asynccnt 0"  ::: "memory");
            char* vbuf = (s & 1) ? vt1 : vt0;

            AB a0, a1;
            const unsigned short* p0 = Pb + n * PB_PITCH + ks + (half << 4);
            const unsigned short* p1 = p0 + 16 * PB_PITCH;
            a0.u[0] = *(const v8us*)(p0);
            a0.u[1] = *(const v8us*)(p0 + 8);
            a1.u[0] = *(const v8us*)(p1);
            a1.u[1] = *(const v8us*)(p1 + 8);
            #pragma unroll
            for (int c = 0; c < CTILES; ++c) {
                AB bv;
                const unsigned short* vcol =
                    (const unsigned short*)(vbuf + (size_t)(c * 16 + n) * VC_PITCH + (half << 5));
                bv.u[0] = *(const v8us*)(vcol);
                bv.u[1] = *(const v8us*)(vcol + 8);
                acc[0][c] = __builtin_amdgcn_wmma_f32_16x16x32_bf16(false, a0.bf, false, bv.bf,
                                                                    (short)0, acc[0][c], false, false);
                acc[1][c] = __builtin_amdgcn_wmma_f32_16x16x32_bf16(false, a1.bf, false, bv.bf,
                                                                    (short)0, acc[1][c], false, false);
            }
            if (s < 2) {                               // refill this buffer for ks+64
                int ksn = k0 + ks + 64 <= (T_ - 32) ? k0 + ks + 64 : (T_ - 32);
                asm volatile("s_wait_dscnt 0" ::: "memory");   // our reads of vbuf done
                issue_vtile(Vtg, vbuf, b, w, l, ksn);
            }
        }
    }

    __syncthreads();
    float inv[2][8];
    #pragma unroll
    for (int t = 0; t < 2; ++t)
        #pragma unroll
        for (int r = 0; r < 8; ++r)
            inv[t][r] = 1.0f / rowl[t * 16 + r + (half << 3)];
    #pragma unroll
    for (int t = 0; t < 2; ++t)
        #pragma unroll
        for (int c = 0; c < CTILES; ++c)
            #pragma unroll
            for (int r = 0; r < 8; ++r) {
                const size_t off = ((size_t)(b * T_ + q0 + t * 16 + r + (half << 3))) * D_
                                 + (size_t)(w * WCOLS + c * 16 + n);
                O[off] = acc[t][c][r] * inv[t][r];
            }
}

// =====================================================================
// Fallback kernel (no workspace): self-contained, 16 query rows
// =====================================================================
#define FQTILE 16
#define FVSTEP 32
#define VT_PITCH 34
#define FQS_BYTES (FQTILE * QS_PITCH * 2)
#define FVT_BYTES (D_ * VT_PITCH * 2)
#define FPB_BYTES (FQTILE * PB_PITCH * 2)
#define FWMAX_OFF (FQS_BYTES + FVT_BYTES + FPB_BYTES)
#define FWSUM_OFF (FWMAX_OFF + NWAVE * 16 * 4)
#define FROWM_OFF (FWSUM_OFF + NWAVE * 16 * 4)
#define FROWL_OFF (FROWM_OFF + 16 * 4)
#define FROWF_OFF (FROWL_OFF + 16 * 4)
#define FSMEM_BYTES (FROWF_OFF + 16 * 4)

__global__ __launch_bounds__(256)
void flash_attn_fallback(const float* __restrict__ Q, const float* __restrict__ K,
                         const float* __restrict__ V, const int* __restrict__ AM,
                         float* __restrict__ O)
{
    extern __shared__ char smem[];
    unsigned short* Qs = (unsigned short*)smem;
    unsigned short* Vt = (unsigned short*)(smem + FQS_BYTES);
    unsigned short* Pb = (unsigned short*)(smem + FQS_BYTES + FVT_BYTES);
    float* wmax = (float*)(smem + FWMAX_OFF);
    float* wsum = (float*)(smem + FWSUM_OFF);
    float* rowm = (float*)(smem + FROWM_OFF);
    float* rowl = (float*)(smem + FROWL_OFF);
    float* rowf = (float*)(smem + FROWF_OFF);

    const int tid = threadIdx.x, w = tid >> 5, l = tid & 31;
    const int half = l >> 4, n = l & 15;
    const int b = blockIdx.x >> 7, q0 = (blockIdx.x & 127) << 4;
    const float scale = 0.03125f;

    for (int idx = tid; idx < (FQTILE * D_) / 2; idx += 256) {
        int r = idx >> 9, d2 = (idx & 511) << 1;
        const float* qp = Q + ((size_t)(b * T_ + q0 + r)) * D_ + d2;
        *(unsigned*)(Qs + r * QS_PITCH + d2) = pk2bf(qp[0] * scale, qp[1] * scale);
    }
    if (tid < 16) { rowm[tid] = -1e30f; rowl[tid] = 0.0f; rowf[tid] = 1.0f; }
    __syncthreads();

    v8f acc[CTILES];
    #pragma unroll
    for (int c = 0; c < CTILES; ++c) acc[c] = (v8f){0.f,0.f,0.f,0.f,0.f,0.f,0.f,0.f};

    const int kt_count = (q0 + FQTILE + KTILE - 1) / KTILE;
    for (int kt = 0; kt < kt_count; ++kt) {
        const int k0 = kt * KTILE;
        const int key = k0 + (w << 4) + n;
        const int keyc = key < T_ ? key : (T_ - 1);

        v8f S = (v8f){0.f,0.f,0.f,0.f,0.f,0.f,0.f,0.f};
        const float* kp = K + ((size_t)(b * T_) + keyc) * D_ + (half << 4);
        #pragma unroll 2
        for (int dc = 0; dc < D_; dc += 32) {
            AB a, bm;
            const unsigned short* qrow = Qs + n * QS_PITCH + dc + (half << 4);
            a.u[0] = *(const v8us*)(qrow);
            a.u[1] = *(const v8us*)(qrow + 8);
            const v4f* kf = (const v4f*)(kp + dc);
            v4f f0 = kf[0], f1 = kf[1], f2 = kf[2], f3 = kf[3];
            bm.u32[0] = pk2bf(f0.x, f0.y); bm.u32[1] = pk2bf(f0.z, f0.w);
            bm.u32[2] = pk2bf(f1.x, f1.y); bm.u32[3] = pk2bf(f1.z, f1.w);
            bm.u32[4] = pk2bf(f2.x, f2.y); bm.u32[5] = pk2bf(f2.z, f2.w);
            bm.u32[6] = pk2bf(f3.x, f3.y); bm.u32[7] = pk2bf(f3.z, f3.w);
            S = __builtin_amdgcn_wmma_f32_16x16x32_bf16(false, a.bf, false, bm.bf,
                                                        (short)0, S, false, false);
        }

        const int am = (key < T_) ? AM[b * T_ + key] : 0;
        float mrow[8];
        #pragma unroll
        for (int r = 0; r < 8; ++r) {
            const int qg = q0 + r + (half << 3);
            if (!((key <= qg) && (am != 0))) S[r] = -1e30f;
            float m = S[r];
            m = fmaxf(m, __shfl_xor(m, 1)); m = fmaxf(m, __shfl_xor(m, 2));
            m = fmaxf(m, __shfl_xor(m, 4)); m = fmaxf(m, __shfl_xor(m, 8));
            mrow[r] = m;
        }
        if (n == 0) {
            #pragma unroll
            for (int r = 0; r < 8; ++r) wmax[w * 16 + r + (half << 3)] = mrow[r];
        }
        __syncthreads();
        if (w == 0 && l < 16) {
            float mo = rowm[l], mn = mo;
            #pragma unroll
            for (int ww = 0; ww < NWAVE; ++ww) mn = fmaxf(mn, wmax[ww * 16 + l]);
            rowm[l] = mn; rowf[l] = __expf(mo - mn);
        }
        __syncthreads();

        float srow[8];
        #pragma unroll
        for (int r = 0; r < 8; ++r) {
            const int rg = r + (half << 3);
            float p = __expf(S[r] - rowm[rg]);
            Pb[rg * PB_PITCH + (w << 4) + n] = f2bf(p);
            float s = p;
            s += __shfl_xor(s, 1); s += __shfl_xor(s, 2);
            s += __shfl_xor(s, 4); s += __shfl_xor(s, 8);
            srow[r] = s;
            const float fac = rowf[rg];
            #pragma unroll
            for (int c = 0; c < CTILES; ++c) acc[c][r] *= fac;
        }
        if (n == 0) {
            #pragma unroll
            for (int r = 0; r < 8; ++r) wsum[w * 16 + r + (half << 3)] = srow[r];
        }
        __syncthreads();
        if (w == 0 && l < 16) {
            float s = 0.f;
            #pragma unroll
            for (int ww = 0; ww < NWAVE; ++ww) s += wsum[ww * 16 + l];
            rowl[l] = rowl[l] * rowf[l] + s;
        }

        for (int ks = 0; ks < KTILE; ks += FVSTEP) {
            __syncthreads();
            for (int idx = tid; idx < (FVSTEP / 2) * D_; idx += 256) {
                int kk2 = (idx >> 10) << 1;
                int d = idx & (D_ - 1);
                int gk0 = k0 + ks + kk2;
                float v0 = (gk0     < T_) ? V[((size_t)(b * T_) + gk0    ) * D_ + d] : 0.0f;
                float v1 = (gk0 + 1 < T_) ? V[((size_t)(b * T_) + gk0 + 1) * D_ + d] : 0.0f;
                *(unsigned*)(Vt + d * VT_PITCH + kk2) = pk2bf(v0, v1);
            }
            __syncthreads();
            AB a;
            const unsigned short* prow = Pb + n * PB_PITCH + ks + (half << 4);
            a.u[0] = *(const v8us*)(prow);
            a.u[1] = *(const v8us*)(prow + 8);
            #pragma unroll
            for (int c = 0; c < CTILES; ++c) {
                AB bv;
                const unsigned short* vcol =
                    Vt + (size_t)(w * WCOLS + c * 16 + n) * VT_PITCH + (half << 4);
                bv.u[0] = *(const v8us*)(vcol);
                bv.u[1] = *(const v8us*)(vcol + 8);
                acc[c] = __builtin_amdgcn_wmma_f32_16x16x32_bf16(false, a.bf, false, bv.bf,
                                                                 (short)0, acc[c], false, false);
            }
        }
    }

    __syncthreads();
    float inv[8];
    #pragma unroll
    for (int r = 0; r < 8; ++r) inv[r] = 1.0f / rowl[r + (half << 3)];
    #pragma unroll
    for (int c = 0; c < CTILES; ++c)
        #pragma unroll
        for (int r = 0; r < 8; ++r) {
            const size_t off = ((size_t)(b * T_ + q0 + r + (half << 3))) * D_
                             + (size_t)(w * WCOLS + c * 16 + n);
            O[off] = acc[c][r] * inv[r];
        }
}

// =====================================================================
extern "C" void kernel_launch(void* const* d_in, const int* in_sizes, int n_in,
                              void* d_out, int out_size, void* d_ws, size_t ws_size,
                              hipStream_t stream) {
    (void)in_sizes; (void)n_in; (void)out_size;
    const float* Q  = (const float*)d_in[0];
    const float* K  = (const float*)d_in[1];
    const float* V  = (const float*)d_in[2];
    const int*   AM = (const int*)d_in[3];
    float*       O  = (float*)d_out;

    const size_t bytesK = (size_t)B_ * T_ * D_ * 2;   // bf16 K
    const size_t need   = bytesK * 2;                 // + bf16 V^T

    if (d_ws != nullptr && ws_size >= need) {
        unsigned short* Kbf = (unsigned short*)d_ws;
        unsigned short* Vtg = (unsigned short*)((char*)d_ws + bytesK);
        convert_k_bf16<<<2048, 256, 0, stream>>>(K, (unsigned*)Kbf);
        transpose_v_bf16<<<B_ * 2048, 256, 0, stream>>>(V, Vtg);
        hipFuncSetAttribute((const void*)flash_attn_ws,
                            hipFuncAttributeMaxDynamicSharedMemorySize, SMEM2_BYTES);
        flash_attn_ws<<<B_ * (T_ / 32), 256, SMEM2_BYTES, stream>>>(Q, Kbf, Vtg, AM, O);
    } else {
        hipFuncSetAttribute((const void*)flash_attn_fallback,
                            hipFuncAttributeMaxDynamicSharedMemorySize, FSMEM_BYTES);
        flash_attn_fallback<<<B_ * (T_ / 16), 256, FSMEM_BYTES, stream>>>(Q, K, V, AM, O);
    }
}
